// Model_61065845015023
// MI455X (gfx1250) — matrix operand order repaired and checked by reading, not compiled, not run on hardware
//
#include <hip/hip_runtime.h>
#include <hip/hip_bf16.h>

// Problem constants (match reference)
#define BB 512
#define TT 128
#define NN 256
#define HH 512

typedef __bf16 bf16;
typedef __attribute__((ext_vector_type(16))) __bf16 v16bf;
typedef __attribute__((ext_vector_type(8)))  __bf16 v8bf;
typedef __attribute__((ext_vector_type(8)))  float   v8f;

static __device__ __forceinline__ v16bf join16(v8bf lo, v8bf hi) {
  return __builtin_shufflevector(lo, hi, 0,1,2,3,4,5,6,7,8,9,10,11,12,13,14,15);
}

// Opaque zero offset in an SGPR: blocks LICM from hoisting the (step-invariant)
// weight-tile loads out of the 128-step loop (round 1 hoisted + spilled them to
// scratch), while keeping the base pointer's global address-space provenance so
// the loads lower to global_load_b128, not flat_load_b128 (flat would couple
// the weight stream into DScnt and falsely serialize against LDS A-fragments).
static __device__ __forceinline__ int opaq0() {
  int z = 0;
  asm volatile("" : "+s"(z));
  return z;
}

// A fragment (16x32 bf16) from row-major [16][lda] buffer (LDS).
// Layout: lanes 0-15 -> K {0..15}; lanes 16-31 -> K {16..31}; M = lane%16.
static __device__ __forceinline__ v16bf load_a(const bf16* base, int lda, int lane, int k0) {
  const bf16* p = base + (lane & 15) * lda + k0 + ((lane >> 4) << 4);
  v8bf lo = *(const v8bf*)p;        // K = k0+off .. +7
  v8bf hi = *(const v8bf*)(p + 8);  // K = k0+off+8 .. +15
  return join16(lo, hi);
}

// B fragment (32x16 bf16) from pre-packed tiles: 1KB/tile, lane-major,
// lane L holds 16 contiguous bf16: N = L%16, K = (L/16)*16 + e.
static __device__ __forceinline__ v16bf load_b(const bf16* tiles, int tileIdx, int lane) {
  const bf16* p = tiles + (size_t)tileIdx * 512 + lane * 16;
  v8bf lo = *(const v8bf*)p;
  v8bf hi = *(const v8bf*)(p + 8);
  return join16(lo, hi);
}

static __device__ __forceinline__ v8f wmma_bf16(v16bf a, v16bf b, v8f c) {
  // (neg_a, A, neg_b, B, c_mod, C, reuse_a, reuse_b)
  return __builtin_amdgcn_wmma_f32_16x16x32_bf16(false, a, false, b, (short)0, c, false, false);
}

static __device__ __forceinline__ float sigf(float x) { return 1.0f / (1.0f + __expf(-x)); }

// ---------------------------------------------------------------------------
// Pack f32 weight matrices into bf16 WMMA B-fragment tiles.
// Logical B matrix is [K][Nout]; tiles ordered nt-major (contiguous K stream per column tile).
// mode 0: W_hr^T  (K=512,N=256)   src0 = W_hr [256,512]
// mode 1: Wf^T    (K=256,N=256)   src0 = W_fr [256,256], zero diagonal
// mode 2: Wwc_m^T (K=256,N=256)   src0 = W_wc [256,512], columns 256..511
// mode 3: [Wih^T; Whh^T] (K=1024,N=2048) src0 = W_ih [2048,512], src1 = W_hh [2048,512]
// ---------------------------------------------------------------------------
__global__ void pack_tiles(const float* __restrict__ s0, const float* __restrict__ s1,
                           bf16* __restrict__ dst, int Kt, int mode, int total) {
  int idx = blockIdx.x * 256 + threadIdx.x;
  if (idx >= total) return;
  int tile = idx >> 9;           // 512 elems / tile
  int r    = idx & 511;
  int lane = r >> 4, e = r & 15;
  int nt = tile / Kt, kt = tile - nt * Kt;
  int n = nt * 16 + (lane & 15);
  int k = kt * 32 + ((lane >> 4) << 4) + e;
  float v;
  if (mode == 0)      v = s0[n * HH + k];
  else if (mode == 1) v = (n == k) ? 0.f : s0[n * NN + k];
  else if (mode == 2) v = s0[n * (2 * NN) + NN + k];
  else                v = (k < 2 * NN) ? s0[n * (2 * NN) + k] : s1[n * HH + (k - 2 * NN)];
  dst[idx] = (bf16)v;
}

// Constant vectors: gamma_h, alpha_const, biases (d_t == 0 in the reference).
__global__ void prep_vec(const float* __restrict__ b_gh, const float* __restrict__ b_gx,
                         const float* __restrict__ W_wc, const float* __restrict__ b_wc,
                         const float* __restrict__ b_hr, const float* __restrict__ b_fr,
                         const float* __restrict__ b_ih, const float* __restrict__ b_hh,
                         float* __restrict__ gH, float* __restrict__ aC,
                         float* __restrict__ vhr, float* __restrict__ vfr, float* __restrict__ bg) {
  int j = blockIdx.x * 256 + threadIdx.x;
  if (j < 4 * HH) bg[j] = b_ih[j] + b_hh[j];
  if (j < HH) gH[j] = __expf(-fmaxf(b_gh[j], 0.f));
  if (j < NN) {
    float s = b_wc[j];
    for (int n = 0; n < NN; ++n) s += W_wc[j * (2 * NN) + n] * __expf(-fmaxf(b_gx[n], 0.f));
    aC[j] = s;
    vhr[j] = b_hr[j];
    vfr[j] = b_fr[j];
  }
}

// ---------------------------------------------------------------------------
// Persistent RNN kernel: 32 blocks x 256 threads (8 waves). Each block owns a
// 16-row batch tile and runs the full T=128 recurrence. Cell state c stays in
// VGPRs; h (pre-scaled by gamma_h, bf16) lives in LDS as part of the gates
// A-matrix [c_c(0..255) | m(256..511) | h(512..1023)].
// ---------------------------------------------------------------------------
__launch_bounds__(256, 1)
__global__ void rits_persistent(const float* __restrict__ x, const float* __restrict__ mask,
                                const bf16* __restrict__ w1, const bf16* __restrict__ w2,
                                const bf16* __restrict__ w3, const bf16* __restrict__ w4,
                                const float* __restrict__ gH, const float* __restrict__ aC,
                                const float* __restrict__ vhr, const float* __restrict__ vfr,
                                const float* __restrict__ bg, float* __restrict__ out) {
  __shared__ __align__(16) bf16 sA[16][1024 + 8];  // padded rows: stagger LDS banks
  __shared__ __align__(16) bf16 sXC[16][256 + 8];
  __shared__ float sGH[HH];
  __shared__ float sAC[NN];
  __shared__ float sBHR[NN];
  __shared__ float sBFR[NN];
  __shared__ float sBG[4 * HH];

  const int tid  = threadIdx.x;
  const int w    = tid >> 5;     // wave 0..7
  const int lane = tid & 31;
  const int b0   = blockIdx.x * 16;
  const int colLo   = lane & 15;
  const int rowBase = (lane >> 4) * 8;   // D-fragment rows: rowBase+v

  for (int i = tid; i < 4 * HH; i += 256) sBG[i] = bg[i];
  for (int i = tid; i < HH; i += 256) sGH[i] = gH[i];
  for (int i = tid; i < NN; i += 256) { sAC[i] = aC[i]; sBHR[i] = vhr[i]; sBFR[i] = vfr[i]; }
  { bf16* f = &sA[0][0]; for (int i = tid; i < 16 * (1024 + 8); i += 256) f[i] = (bf16)0.f; }

  v8f cst[4] = {};  // LSTM cell state: wave's 64 H-columns x 16 rows, in registers

  __syncthreads();

#pragma clang loop unroll(disable)
  for (int t = 0; t < TT; ++t) {
    // ---- P0: load x,m fragments for this wave's two N-column tiles; stage m (bf16)
    float mfr[2][8], xfr[2][8];
    #pragma unroll
    for (int ct = 0; ct < 2; ++ct) {
      const int col = (w * 2 + ct) * 16 + colLo;
      #pragma unroll
      for (int v = 0; v < 8; ++v) {
        const int row = rowBase + v;
        const size_t gidx = ((size_t)(b0 + row) * TT + t) * NN + col;
        xfr[ct][v] = x[gidx];
        mfr[ct][v] = mask[gidx];
        sA[row][NN + col] = (bf16)mfr[ct][v];
        if (t + 1 < TT) __builtin_prefetch(&x[gidx + NN], 0, 1);  // global_prefetch_b8, next step
      }
    }
    __syncthreads();  // barrier A: x/m staged, h(t-1) visible

    // ---- P1: x_h = h @ W_hr^T + b_hr   (K = 512, from sA cols 512..1023)
    v8f xh[2] = {};
    {
      const bf16* w1t = w1 + opaq0();  // defeat cross-step LICM, keep global AS
      for (int kt = 0; kt < 16; ++kt) {
        v16bf a = load_a(&sA[0][0], 1024 + 8, lane, 2 * NN + kt * 32);
        xh[0] = wmma_bf16(a, load_b(w1t, (w * 2 + 0) * 16 + kt, lane), xh[0]);
        xh[1] = wmma_bf16(a, load_b(w1t, (w * 2 + 1) * 16 + kt, lane), xh[1]);
      }
    }
    #pragma unroll
    for (int ct = 0; ct < 2; ++ct) {   // x_c = m*x + (1-m)*x_h -> bf16 stage
      const int col = (w * 2 + ct) * 16 + colLo;
      const float bias = sBHR[col];
      #pragma unroll
      for (int v = 0; v < 8; ++v) {
        xh[ct][v] += bias;
        float xc = mfr[ct][v] * xfr[ct][v] + (1.f - mfr[ct][v]) * xh[ct][v];
        sXC[rowBase + v][col] = (bf16)xc;
      }
    }
    __syncthreads();  // barrier B: x_c staged

    // ---- P2: z_h = x_c @ Wf^T + b_fr ;  alpha = m @ Wwc_m^T + alpha_const  (K = 256)
    v8f zf[2] = {}, al[2] = {};
    {
      const bf16* w2t = w2 + opaq0();
      const bf16* w3t = w3 + opaq0();
      for (int kt = 0; kt < 8; ++kt) {
        v16bf a2 = load_a(&sXC[0][0], 256 + 8, lane, kt * 32);
        v16bf a3 = load_a(&sA[0][0], 1024 + 8, lane, NN + kt * 32);
        #pragma unroll
        for (int ct = 0; ct < 2; ++ct) {
          const int ti = (w * 2 + ct) * 8 + kt;
          zf[ct] = wmma_bf16(a2, load_b(w2t, ti, lane), zf[ct]);
          al[ct] = wmma_bf16(a3, load_b(w3t, ti, lane), al[ct]);
        }
      }
    }
    #pragma unroll
    for (int ct = 0; ct < 2; ++ct) {   // c_h, c_c; c_c -> output and gates A-matrix
      const int col = (w * 2 + ct) * 16 + colLo;
      const float bfr_ = sBFR[col], ac_ = sAC[col];
      #pragma unroll
      for (int v = 0; v < 8; ++v) {
        const int row = rowBase + v;
        float z  = zf[ct][v] + bfr_;
        float a  = al[ct][v] + ac_;
        float ch = a * z + (1.f - a) * xh[ct][v];
        float cc = mfr[ct][v] * xfr[ct][v] + (1.f - mfr[ct][v]) * ch;
        sA[row][col] = (bf16)cc;
        out[((size_t)(b0 + row) * TT + t) * NN + col] = cc;
      }
    }
    __syncthreads();  // barrier C: full gates A-matrix [c_c|m|h] ready

    // ---- P3: gates = A @ [Wih^T;Whh^T] + b   (K = 1024, wave owns 4 col-tiles per gate)
    v8f acc[4][4] = {};
    {
      const bf16* w4t = w4 + opaq0();
      for (int kt = 0; kt < 32; ++kt) {
        v16bf a = load_a(&sA[0][0], 1024 + 8, lane, kt * 32);
        #pragma unroll
        for (int g = 0; g < 4; ++g)
          #pragma unroll
          for (int j = 0; j < 4; ++j) {
            const int nt = g * 32 + w * 4 + j;
            acc[g][j] = wmma_bf16(a, load_b(w4t, nt * 32 + kt, lane), acc[g][j]);
          }
      }
    }
    __syncthreads();  // barrier D: all sA reads done before h is overwritten

    #pragma unroll
    for (int j = 0; j < 4; ++j) {       // LSTM cell update, all in registers
      const int hcol = w * 64 + j * 16 + colLo;
      const float bi = sBG[0 * HH + hcol], bf = sBG[1 * HH + hcol];
      const float bgv = sBG[2 * HH + hcol], bo = sBG[3 * HH + hcol];
      const float gam = sGH[hcol];
      #pragma unroll
      for (int v = 0; v < 8; ++v) {
        float iv = acc[0][j][v] + bi;
        float fv = acc[1][j][v] + bf;
        float gv = acc[2][j][v] + bgv;
        float ov = acc[3][j][v] + bo;
        float c  = sigf(fv) * cst[j][v] + sigf(iv) * tanhf(gv);
        float h  = sigf(ov) * tanhf(c);
        cst[j][v] = c;
        sA[rowBase + v][2 * NN + hcol] = (bf16)(h * gam);  // pre-decay for step t+1
      }
    }
    // barrier A of next iteration publishes h
  }
}

extern "C" void kernel_launch(void* const* d_in, const int* in_sizes, int n_in,
                              void* d_out, int out_size, void* d_ws, size_t ws_size,
                              hipStream_t stream) {
  const float* x    = (const float*)d_in[0];
  const float* mask = (const float*)d_in[4];
  const float* b_gh = (const float*)d_in[6];
  const float* b_gx = (const float*)d_in[8];
  const float* W_hr = (const float*)d_in[9];
  const float* b_hr = (const float*)d_in[10];
  const float* W_fr = (const float*)d_in[11];
  const float* b_fr = (const float*)d_in[12];
  const float* W_wc = (const float*)d_in[13];
  const float* b_wc = (const float*)d_in[14];
  const float* W_ih = (const float*)d_in[15];
  const float* W_hh = (const float*)d_in[16];
  const float* b_ih = (const float*)d_in[17];
  const float* b_hh = (const float*)d_in[18];
  float* out = (float*)d_out;

  char* ws = (char*)d_ws;
  bf16* w1 = (bf16*)(ws);                          // 256 KB : W_hr^T tiles
  bf16* w2 = (bf16*)(ws + (256 << 10));            // 128 KB : Wf^T tiles
  bf16* w3 = (bf16*)(ws + (384 << 10));            // 128 KB : Wwc_m^T tiles
  bf16* w4 = (bf16*)(ws + (512 << 10));            // 4 MB   : [Wih^T;Whh^T] tiles
  float* gH  = (float*)(ws + (512 << 10) + (4 << 20));
  float* aC  = gH + 512;
  float* vhr = aC + 256;
  float* vfr = vhr + 256;
  float* bg  = vfr + 256;

  pack_tiles<<<(131072 + 255) / 256, 256, 0, stream>>>(W_hr, nullptr, w1, 16, 0, 131072);
  pack_tiles<<<(65536 + 255) / 256, 256, 0, stream>>>(W_fr, nullptr, w2, 8, 1, 65536);
  pack_tiles<<<(65536 + 255) / 256, 256, 0, stream>>>(W_wc, nullptr, w3, 8, 2, 65536);
  pack_tiles<<<(2097152 + 255) / 256, 256, 0, stream>>>(W_ih, W_hh, w4, 32, 3, 2097152);
  prep_vec<<<8, 256, 0, stream>>>(b_gh, b_gx, W_wc, b_wc, b_hr, b_fr, b_ih, b_hh,
                                  gH, aC, vhr, vfr, bg);
  rits_persistent<<<32, 256, 0, stream>>>(x, mask, w1, w2, w3, w4, gH, aC, vhr, vfr, bg, out);
  (void)in_sizes; (void)n_in; (void)out_size; (void)ws_size;
}
